// SequentialPredictor_11708080849548
// MI455X (gfx1250) — matrix operand order repaired and checked by reading, not compiled, not run on hardware
//
#include <hip/hip_runtime.h>

// SequentialPredictor: B=32768, T=8, OUT_LEN=12, H=64 (4H=256)
// One wave32 handles 16 batch rows; all matmuls via v_wmma_f32_16x16x32_f16.

typedef __attribute__((ext_vector_type(16))) _Float16 v16h;
typedef __attribute__((ext_vector_type(8)))  float    v8f;

#define NB      32768
#define TT      8
#define OUTLEN  12
#define WPB     8              // waves per block
#define ROWS_PB (WPB * 16)     // 128 rows per block

__device__ __forceinline__ v8f wmma16(v16h a, v16h b, v8f c) {
  // D = A(16x32 f16) * B(32x16 f16) + C(16x16 f32)
  return __builtin_amdgcn_wmma_f32_16x16x32_f16(false, a, false, b, (short)0, c,
                                                false, false);
}

__device__ __forceinline__ float sigm(float x) {
  float e = __expf(-x);
  return __builtin_amdgcn_rcpf(1.0f + e);
}
__device__ __forceinline__ float tanha(float x) {
  float ax = fabsf(x);
  float e  = __expf(-2.0f * ax);
  float r  = (1.0f - e) * __builtin_amdgcn_rcpf(1.0f + e);
  return copysignf(r, x);
}
__device__ __forceinline__ float leaky(float v) { return v > 0.0f ? v : 0.1f * v; }

// Fill a B-operand image in LDS for Y = A @ W.T (W row-major [N][K], K=64).
// CDNA5 16-bit B 32x16 layout: lane = N (lo/hi half-wave), element e -> K = (e&7) + 16*(e>=8) + 8*(lane>=16).
// Image index: ((j*2 + kh)*32 + lane)*16 + e, kh selects K-half (0..31 / 32..63).
static __device__ __forceinline__ void fillB(const float* __restrict__ W,
                                             _Float16* __restrict__ dst,
                                             int ntiles, int nvalid, int tid) {
  int total = ntiles << 10;  // ntiles * 2 * 32 * 16
  for (int idx = tid; idx < total; idx += 256) {
    int e  = idx & 15;
    int ln = (idx >> 4) & 31;
    int kh = (idx >> 9) & 1;
    int j  = idx >> 10;
    int k  = kh * 32 + (e & 7) + ((e & 8) ? 16 : 0) + ((ln & 16) ? 8 : 0);
    int n  = 16 * j + (ln & 15);
    dst[idx] = (_Float16)((n < nvalid) ? W[n * 64 + k] : 0.0f);
  }
}

__global__ __launch_bounds__(256) void seqpred_kernel(
    const float* __restrict__ x,
    const float* __restrict__ W_ih,  const float* __restrict__ W_hh,
    const float* __restrict__ b_ih,  const float* __restrict__ b_hh,
    const float* __restrict__ fc1_w1, const float* __restrict__ fc1_b1,
    const float* __restrict__ fc1_w2, const float* __restrict__ fc1_b2,
    const float* __restrict__ fo_w1,  const float* __restrict__ fo_b1,
    const float* __restrict__ fo_w2,  const float* __restrict__ fo_b2,
    float* __restrict__ out) {
  extern __shared__ char smem[];
  _Float16* whhF = (_Float16*)smem;        // 16 tiles  -> 16384 halves
  _Float16* w1F  = whhF + 16384;           // 4 tiles   ->  4096
  _Float16* w2F  = w1F + 4096;             // 4 tiles
  _Float16* w3F  = w2F + 4096;             // 4 tiles
  _Float16* w4F  = w3F + 4096;             // 1 tile (padded N=2) -> 1024
  _Float16* imgs = w4F + 1024;             // per-wave hA/aA/aB: 3*1024 halves each
  float*    xba  = (float*)(imgs + WPB * 3072);  // per-wave x buffer: 256 floats

  const int tid  = threadIdx.x;
  const int lane = tid & 31;
  const int wv   = tid >> 5;
  const int nlo  = lane & 15;
  const int hi16 = lane >> 4;  // 0: rows 0..7 of D tile, 1: rows 8..15

  // ---- cooperative weight staging (f32 -> f16, B-operand layout) ----
  fillB(W_hh,   whhF, 16, 256, tid);
  fillB(fc1_w1, w1F,  4,  64,  tid);
  fillB(fc1_w2, w2F,  4,  64,  tid);
  fillB(fo_w1,  w3F,  4,  64,  tid);
  fillB(fo_w2,  w4F,  1,  2,   tid);

  // ---- per-wave staging ----
  _Float16* hA = imgs + wv * 3072;  // LSTM hidden state, A-operand image (16x64 f16)
  _Float16* aA = hA + 1024;         // FC activation ping
  _Float16* aB = aA + 1024;         // FC activation pong
  float*    xb = xba + (wv << 8);   // [16 rows][8 t][2] floats; slot t=7 carries `last`
  const int rowBase = blockIdx.x * ROWS_PB + wv * 16;

  for (int i = lane; i < 1024; i += 32) hA[i] = (_Float16)0.0f;  // h0 = 0
  for (int i = lane; i < 256; i += 32)
    xb[i] = x[(size_t)(rowBase + (i >> 4)) * 16 + (i & 15)];

  __syncthreads();

  // ---- per-lane constants: column n depends only on (tile j, nlo) ----
  float bv[16], wa[16], wb[16];
#pragma unroll
  for (int j = 0; j < 16; ++j) {
    int n = 16 * j + nlo;
    bv[j] = b_ih[n] + b_hh[n];
    wa[j] = W_ih[2 * n];
    wb[j] = W_ih[2 * n + 1];
  }
  float fb1[4], fb2[4], fb3[4];
#pragma unroll
  for (int jj = 0; jj < 4; ++jj) {
    fb1[jj] = fc1_b1[16 * jj + nlo];
    fb2[jj] = fc1_b2[16 * jj + nlo];
    fb3[jj] = fo_b1[16 * jj + nlo];
  }
  const float fob2 = fo_b2[nlo < 2 ? nlo : 1];

  // D-tile -> A-image scatter constants: element (M, k=16*jj+nlo)
  const int la = (nlo >= 8) ? 16 : 0;  // K sub-block 8..15 / 24..31 lives in lanes 16..31
  const int eb = (nlo & 7);

  float cst[32];  // cell state c, D layout, f32
#pragma unroll
  for (int i = 0; i < 32; ++i) cst[i] = 0.0f;

  for (int outer = 0; outer < OUTLEN; ++outer) {
    // ================= 8 LSTM steps =================
    for (int t = 0; t < TT; ++t) {
      asm volatile("s_wait_dscnt 0x0" ::: "memory");  // hA scatter -> gather
      v16h A0 = *(const v16h*)(hA + lane * 16);
      v16h A1 = *(const v16h*)(hA + 512 + lane * 16);
      float xv0[8], xv1[8];
#pragma unroll
      for (int r = 0; r < 8; ++r) {
        int M  = r + 8 * hi16;
        xv0[r] = xb[M * 16 + t * 2];
        xv1[r] = xb[M * 16 + t * 2 + 1];
      }
#pragma unroll
      for (int jj = 0; jj < 4; ++jj) {  // column group: gates i/f/g/o = tiles jj,+4,+8,+12
        v8f zi, zf, zg, zo;
#pragma unroll
        for (int r = 0; r < 8; ++r) {  // acc init = bias + xt @ W_ih.T (K=2, VALU)
          zi[r] = bv[jj]      + xv0[r] * wa[jj]      + xv1[r] * wb[jj];
          zf[r] = bv[jj + 4]  + xv0[r] * wa[jj + 4]  + xv1[r] * wb[jj + 4];
          zg[r] = bv[jj + 8]  + xv0[r] * wa[jj + 8]  + xv1[r] * wb[jj + 8];
          zo[r] = bv[jj + 12] + xv0[r] * wa[jj + 12] + xv1[r] * wb[jj + 12];
        }
        {
          int j = jj;
          v16h b0 = *(const v16h*)(whhF + ((j * 2 + 0) * 32 + lane) * 16);
          v16h b1 = *(const v16h*)(whhF + ((j * 2 + 1) * 32 + lane) * 16);
          zi = wmma16(A0, b0, zi); zi = wmma16(A1, b1, zi);
        }
        {
          int j = jj + 4;
          v16h b0 = *(const v16h*)(whhF + ((j * 2 + 0) * 32 + lane) * 16);
          v16h b1 = *(const v16h*)(whhF + ((j * 2 + 1) * 32 + lane) * 16);
          zf = wmma16(A0, b0, zf); zf = wmma16(A1, b1, zf);
        }
        {
          int j = jj + 8;
          v16h b0 = *(const v16h*)(whhF + ((j * 2 + 0) * 32 + lane) * 16);
          v16h b1 = *(const v16h*)(whhF + ((j * 2 + 1) * 32 + lane) * 16);
          zg = wmma16(A0, b0, zg); zg = wmma16(A1, b1, zg);
        }
        {
          int j = jj + 12;
          v16h b0 = *(const v16h*)(whhF + ((j * 2 + 0) * 32 + lane) * 16);
          v16h b1 = *(const v16h*)(whhF + ((j * 2 + 1) * 32 + lane) * 16);
          zo = wmma16(A0, b0, zo); zo = wmma16(A1, b1, zo);
        }
#pragma unroll
        for (int r = 0; r < 8; ++r) {
          float si = sigm(zi[r]);
          float sf = sigm(zf[r]);
          float gg = tanha(zg[r]);
          float so = sigm(zo[r]);
          float cN = sf * cst[jj * 8 + r] + si * gg;
          cst[jj * 8 + r] = cN;
          float hv = so * tanha(cN);
          int M = r + 8 * hi16;
          hA[(jj >> 1) * 512 + (M + la) * 16 + eb + ((jj & 1) ? 8 : 0)] = (_Float16)hv;
        }
      }
    }

    // ================= FC head =================
    // L1: aA = leaky(h @ fc1_w1.T + fc1_b1)
    asm volatile("s_wait_dscnt 0x0" ::: "memory");
    {
      v16h A0 = *(const v16h*)(hA + lane * 16);
      v16h A1 = *(const v16h*)(hA + 512 + lane * 16);
#pragma unroll
      for (int jj = 0; jj < 4; ++jj) {
        v8f acc;
#pragma unroll
        for (int r = 0; r < 8; ++r) acc[r] = fb1[jj];
        v16h b0 = *(const v16h*)(w1F + ((jj * 2 + 0) * 32 + lane) * 16);
        v16h b1 = *(const v16h*)(w1F + ((jj * 2 + 1) * 32 + lane) * 16);
        acc = wmma16(A0, b0, acc); acc = wmma16(A1, b1, acc);
#pragma unroll
        for (int r = 0; r < 8; ++r) {
          int M = r + 8 * hi16;
          aA[(jj >> 1) * 512 + (M + la) * 16 + eb + ((jj & 1) ? 8 : 0)] =
              (_Float16)leaky(acc[r]);
        }
      }
    }
    // L2: aB = aA @ fc1_w2.T + fc1_b2
    asm volatile("s_wait_dscnt 0x0" ::: "memory");
    {
      v16h A0 = *(const v16h*)(aA + lane * 16);
      v16h A1 = *(const v16h*)(aA + 512 + lane * 16);
#pragma unroll
      for (int jj = 0; jj < 4; ++jj) {
        v8f acc;
#pragma unroll
        for (int r = 0; r < 8; ++r) acc[r] = fb2[jj];
        v16h b0 = *(const v16h*)(w2F + ((jj * 2 + 0) * 32 + lane) * 16);
        v16h b1 = *(const v16h*)(w2F + ((jj * 2 + 1) * 32 + lane) * 16);
        acc = wmma16(A0, b0, acc); acc = wmma16(A1, b1, acc);
#pragma unroll
        for (int r = 0; r < 8; ++r) {
          int M = r + 8 * hi16;
          aB[(jj >> 1) * 512 + (M + la) * 16 + eb + ((jj & 1) ? 8 : 0)] =
              (_Float16)acc[r];
        }
      }
    }
    // L3: aA = leaky(aB @ fo_w1.T + fo_b1)
    asm volatile("s_wait_dscnt 0x0" ::: "memory");
    {
      v16h A0 = *(const v16h*)(aB + lane * 16);
      v16h A1 = *(const v16h*)(aB + 512 + lane * 16);
#pragma unroll
      for (int jj = 0; jj < 4; ++jj) {
        v8f acc;
#pragma unroll
        for (int r = 0; r < 8; ++r) acc[r] = fb3[jj];
        v16h b0 = *(const v16h*)(w3F + ((jj * 2 + 0) * 32 + lane) * 16);
        v16h b1 = *(const v16h*)(w3F + ((jj * 2 + 1) * 32 + lane) * 16);
        acc = wmma16(A0, b0, acc); acc = wmma16(A1, b1, acc);
#pragma unroll
        for (int r = 0; r < 8; ++r) {
          int M = r + 8 * hi16;
          aA[(jj >> 1) * 512 + (M + la) * 16 + eb + ((jj & 1) ? 8 : 0)] =
              (_Float16)leaky(acc[r]);
        }
      }
    }
    // L4: y = aA @ fo_w2.T + fo_b2  (one padded 16-wide tile; n<2 valid)
    asm volatile("s_wait_dscnt 0x0" ::: "memory");
    {
      v16h A0 = *(const v16h*)(aA + lane * 16);
      v16h A1 = *(const v16h*)(aA + 512 + lane * 16);
      v8f acc;
#pragma unroll
      for (int r = 0; r < 8; ++r) acc[r] = 0.0f;
      v16h b0 = *(const v16h*)(w4F + (0 * 32 + lane) * 16);
      v16h b1 = *(const v16h*)(w4F + (1 * 32 + lane) * 16);
      acc = wmma16(A0, b0, acc); acc = wmma16(A1, b1, acc);
      if (nlo < 2) {
#pragma unroll
        for (int r = 0; r < 8; ++r) {
          int M = r + 8 * hi16;
          float yv = acc[r] + fob2;
          out[(size_t)(rowBase + M) * (OUTLEN * 2) + outer * 2 + nlo] = yv;
          int xo = M * 16 + 14 + nlo;  // slot t=7 : last += y
          xb[xo] = xb[xo] + yv;
        }
      }
    }
  }
}

extern "C" void kernel_launch(void* const* d_in, const int* in_sizes, int n_in,
                              void* d_out, int out_size, void* d_ws, size_t ws_size,
                              hipStream_t stream) {
  (void)in_sizes; (void)n_in; (void)out_size; (void)d_ws; (void)ws_size;
  const float* x      = (const float*)d_in[0];
  // d_in[1] = noise (unused by reference output)
  const float* W_ih   = (const float*)d_in[2];
  const float* W_hh   = (const float*)d_in[3];
  const float* b_ih   = (const float*)d_in[4];
  const float* b_hh   = (const float*)d_in[5];
  const float* fc1_w1 = (const float*)d_in[6];
  const float* fc1_b1 = (const float*)d_in[7];
  const float* fc1_w2 = (const float*)d_in[8];
  const float* fc1_b2 = (const float*)d_in[9];
  const float* fo_w1  = (const float*)d_in[10];
  const float* fo_b1  = (const float*)d_in[11];
  const float* fo_w2  = (const float*)d_in[12];
  const float* fo_b2  = (const float*)d_in[13];
  float* out = (float*)d_out;

  // LDS: weight images (16+4+4+4+1 tiles) + per-wave hA/aA/aB + x buffers
  constexpr size_t SHMEM =
      (size_t)(16384 + 3 * 4096 + 1024 + WPB * 3072) * 2 + (size_t)WPB * 256 * 4;
  static bool attr_set = false;
  if (!attr_set) {
    (void)hipFuncSetAttribute((const void*)seqpred_kernel,
                              hipFuncAttributeMaxDynamicSharedMemorySize,
                              (int)SHMEM);
    attr_set = true;
  }
  dim3 grid(NB / ROWS_PB);   // 256 blocks
  dim3 block(WPB * 32);      // 256 threads = 8 waves
  seqpred_kernel<<<grid, block, SHMEM, stream>>>(
      x, W_ih, W_hh, b_ih, b_hh, fc1_w1, fc1_b1, fc1_w2, fc1_b2,
      fo_w1, fo_b1, fo_w2, fo_b2, out);
}
